// CognitiveAttention_54503134986904
// MI455X (gfx1250) — matrix operand order repaired and checked by reading, not compiled, not run on hardware
//
#include <hip/hip_runtime.h>
#include <hip/hip_bf16.h>
#include <cstddef>

// ---------------------------------------------------------------------------
// Types for CDNA5 WMMA (wave32): D(16x16 f32) = A(16x32 bf16) x B(32x16 bf16) + C
// ---------------------------------------------------------------------------
typedef __attribute__((ext_vector_type(16))) __bf16 v16bf;
typedef __attribute__((ext_vector_type(8)))  __bf16 v8bf;
typedef __attribute__((ext_vector_type(4)))  __bf16 v4bf;
typedef __attribute__((ext_vector_type(8)))  float  v8f;
typedef __attribute__((ext_vector_type(4)))  unsigned int v4u;
typedef __attribute__((ext_vector_type(4)))  int    v4i;
typedef __attribute__((ext_vector_type(8)))  int    v8i;

union FragBF {
    v16bf v;
    v8bf  h[2];
};

__device__ __forceinline__ v8f zero8() {
    v8f z = {0.f, 0.f, 0.f, 0.f, 0.f, 0.f, 0.f, 0.f};
    return z;
}

__device__ __forceinline__ v8f wmma_bf16(v16bf a, v16bf b, v8f c) {
    return __builtin_amdgcn_wmma_f32_16x16x32_bf16(
        /*neg_a=*/false, a, /*neg_b=*/false, b,
        /*c_mod=*/(short)0, c, /*reuse_a=*/false, /*reuse_b=*/false);
}

// ---------------------------------------------------------------------------
// TDM: DMA a [rows x 64] bf16 tile (row stride ld_elems in memory) into LDS
// laid out with a 72-half (144 B) row stride: 32 DWORDs data + 4 DWORDs pad.
// D# group0/group1 packing per CDNA5 ISA 8.3/8.4. 2D tile, no multicast.
// Issued by one wave; tracked by TENSORcnt.
// ---------------------------------------------------------------------------
__device__ __forceinline__ void tdm_load_tile_bf16(const __bf16* gsrc,
                                                   unsigned lds_byte_off,
                                                   int ld_elems, int rows,
                                                   int tensor_rows) {
    unsigned long long ga = (unsigned long long)(size_t)gsrc;
    v4u g0;
    g0[0] = 1u;                                            // count=1, user desc
    g0[1] = lds_byte_off;                                  // lds_addr
    g0[2] = (unsigned)(ga & 0xFFFFFFFFull);                // global_addr[31:0]
    g0[3] = (unsigned)((ga >> 32) & 0x1FFFFFFull)          // global_addr[56:32]
          | (2u << 30);                                    // type = 2 ("image")
    v8i g1;
    g1[0] = (1 << 16)     // data_size = 1 -> 2 bytes
          | (1 << 20)     // pad_enable
          | (4 << 22)     // pad_interval = 32 DWORDs (one 64-elem bf16 row)
          | (3 << 25);    // pad_amount  = 4 DWORDs  (8 halves -> 72 stride)
    g1[1] = (ld_elems & 0xFFFF) << 16;                     // tensor_dim0[15:0]
    g1[2] = ((unsigned)ld_elems >> 16)                     // tensor_dim0[31:16]
          | ((tensor_rows & 0xFFFF) << 16);                // tensor_dim1[15:0]
    g1[3] = ((unsigned)tensor_rows >> 16)                  // tensor_dim1[31:16]
          | (64 << 16);                                    // tile_dim0 = 64
    g1[4] = rows;                                          // tile_dim1, tile_dim2=0
    g1[5] = ld_elems;                                      // tensor_dim0_stride[31:0]
    g1[6] = 0;                                             // stride hi, dim1_stride lo
    g1[7] = 0;
    v4i z4 = {0, 0, 0, 0};
#if defined(__clang_major__) && (__clang_major__ >= 23)
    v8i z8 = {0, 0, 0, 0, 0, 0, 0, 0};
    __builtin_amdgcn_tensor_load_to_lds(g0, g1, z4, z4, z8, 0);
#else
    __builtin_amdgcn_tensor_load_to_lds(g0, g1, z4, z4, 0);
#endif
}

// ---------------------------------------------------------------------------
// Fragment loads from LDS (stride 72 halves = 64 data + 8 pad).
//
// A-matrix 16x32 bf16 (ISA 7.12.2): lanes 0-15 = rows, hold K=0..7 & 16..23;
// lanes 16-31 hold K=8..15 & 24..31 of the same rows.
// B-matrix 32x16 bf16: lane n (0-15) holds K=0..7 & 16..23 of column n; lane n+16
// holds K=8..15 & 24..31 of column n. All chunks are 16-byte contiguous.
// ---------------------------------------------------------------------------
__device__ __forceinline__ v16bf load_afrag_lds(const __bf16 (*lds)[72], int mrow,
                                                int kc, int lane) {
    int m = mrow + (lane & 15);
    int half = lane >> 4;
    FragBF u;
    u.h[0] = *(const v8bf*)&lds[m][kc * 32 + half * 8];
    u.h[1] = *(const v8bf*)&lds[m][kc * 32 + 16 + half * 8];
    return u.v;
}

__device__ __forceinline__ v16bf load_bfrag_lds(const __bf16 (*lds)[72], int nrow,
                                                int kc, int lane) {
    int n = nrow + (lane & 15);
    int half = lane >> 4;
    FragBF u;
    u.h[0] = *(const v8bf*)&lds[n][kc * 32 + half * 8];
    u.h[1] = *(const v8bf*)&lds[n][kc * 32 + 16 + half * 8];
    return u.v;
}

// ---------------------------------------------------------------------------
// Generic NT GEMM:  C[M,N] = A[M,K] * W[N,K]^T
//   A_F32:   A fp32 (VALU-converted to bf16 on LDS stage); else bf16 via TDM
//   OUT_F32: C fp32 else bf16
//   TRANS_OUT: store transposed per batch: out[(b*N + n)*S + s], m = b*S+s
// Block tile 128x128, 8 waves of 32x64, K-step 64. M,N % 128 == 0, K % 64 == 0.
// ---------------------------------------------------------------------------
template <bool A_F32, bool OUT_F32, bool TRANS_OUT>
__global__ __launch_bounds__(256)
void gemm_nt_kernel(const void* __restrict__ Av, const float* __restrict__ W,
                    void* __restrict__ Cv, int M, int N, int K, int S) {
    __shared__ __bf16 ldsA[128][72];
    __shared__ __bf16 ldsB[128][72];

    const int tid  = threadIdx.x;
    const int w    = tid >> 5;
    const int lane = tid & 31;
    const int half = lane >> 4;
    const int nh   = lane & 15;

    const int m0 = blockIdx.y * 128;
    const int n0 = blockIdx.x * 128;
    const int wm = (w >> 1) * 32;   // wave M offset: 0,32,64,96
    const int wn = (w & 1) * 64;    // wave N offset: 0,64

    v8f acc[2][4];
#pragma unroll
    for (int mi = 0; mi < 2; ++mi)
#pragma unroll
        for (int ni = 0; ni < 4; ++ni) acc[mi][ni] = zero8();

    const float*  Af = (const float*)Av;
    const __bf16* Ab = (const __bf16*)Av;
    const unsigned ldsA_off = (unsigned)(size_t)(void*)&ldsA[0][0];

    for (int k0 = 0; k0 < K; k0 += 64) {
        __syncthreads();
        // ---- A tile: 128 x 64 ----
        if constexpr (A_F32) {
#pragma unroll
            for (int i = 0; i < 8; ++i) {        // 2048 float4 units / 256 threads
                int t4 = i * 256 + tid;
                int r  = t4 >> 4;
                int c4 = t4 & 15;
                float4 f = *(const float4*)(Af + (size_t)(m0 + r) * K + k0 + c4 * 4);
                v4bf hq;
                hq[0] = (__bf16)f.x; hq[1] = (__bf16)f.y;
                hq[2] = (__bf16)f.z; hq[3] = (__bf16)f.w;
                *(v4bf*)&ldsA[r][c4 * 4] = hq;
            }
        } else {
            if (w == 0) {
                tdm_load_tile_bf16(Ab + (size_t)m0 * K + k0, ldsA_off, K, 128, M);
            }
        }
        // ---- W tile (always fp32 -> bf16): 128 x 64 ----
#pragma unroll
        for (int i = 0; i < 8; ++i) {
            int t4 = i * 256 + tid;
            int r  = t4 >> 4;
            int c4 = t4 & 15;
            float4 f = *(const float4*)(W + (size_t)(n0 + r) * K + k0 + c4 * 4);
            v4bf hq;
            hq[0] = (__bf16)f.x; hq[1] = (__bf16)f.y;
            hq[2] = (__bf16)f.z; hq[3] = (__bf16)f.w;
            *(v4bf*)&ldsB[r][c4 * 4] = hq;
        }
        if constexpr (!A_F32) {
            if (w == 0) __builtin_amdgcn_s_wait_tensorcnt(0);
        }
        __syncthreads();

#pragma unroll
        for (int kc = 0; kc < 2; ++kc) {
            v16bf am[2];
#pragma unroll
            for (int mi = 0; mi < 2; ++mi)
                am[mi] = load_afrag_lds(ldsA, wm + mi * 16, kc, lane);
            v16bf bn[4];
#pragma unroll
            for (int ni = 0; ni < 4; ++ni)
                bn[ni] = load_bfrag_lds(ldsB, wn + ni * 16, kc, lane);
#pragma unroll
            for (int mi = 0; mi < 2; ++mi)
#pragma unroll
                for (int ni = 0; ni < 4; ++ni)
                    acc[mi][ni] = wmma_bf16(am[mi], bn[ni], acc[mi][ni]);
        }
    }

    // ---- store: C layout = VGPR r, lanes 0-15 -> row r, lanes 16-31 -> row 8+r
    float*  Cf = (float*)Cv;
    __bf16* Cb = (__bf16*)Cv;
#pragma unroll
    for (int mi = 0; mi < 2; ++mi) {
#pragma unroll
        for (int ni = 0; ni < 4; ++ni) {
#pragma unroll
            for (int r = 0; r < 8; ++r) {
                int row = m0 + wm + mi * 16 + half * 8 + r;
                int col = n0 + wn + ni * 16 + nh;
                float val = acc[mi][ni][r];
                if constexpr (TRANS_OUT) {
                    int bb = row / S;
                    int s  = row - bb * S;
                    Cb[((size_t)bb * N + col) * S + s] = (__bf16)val;
                } else if constexpr (OUT_F32) {
                    Cf[(size_t)row * N + col] = val;
                } else {
                    Cb[(size_t)row * N + col] = (__bf16)val;
                }
            }
        }
    }
}

// ---------------------------------------------------------------------------
// Flash attention: grid = B*H*(S/128), block = 256 (8 waves), each wave owns
// 16 query rows. Streams keys in chunks of 32 with online softmax.
//   q   : bf16 [B*S, H*HD]
//   k   : bf16 [B*S, HKV*HD]
//   vT  : bf16 [B, HKV*HD, S]   (transposed: d-major rows, key-contiguous)
//   mask: f32  [B, 1, S, S]
//   ctx : bf16 [B*S, H*HD]
// ---------------------------------------------------------------------------
__global__ __launch_bounds__(256)
void attn_kernel(const __bf16* __restrict__ q, const __bf16* __restrict__ k,
                 const __bf16* __restrict__ vT, const float* __restrict__ mask,
                 __bf16* __restrict__ ctx) {
    constexpr int Hc = 32, HKVc = 8, HDc = 64, Sc = 2048;
    constexpr int QW = Hc * HDc;     // 2048
    constexpr int KW = HKVc * HDc;   // 512
    const float scale = 0.125f;      // 1/sqrt(64)

    __shared__ __bf16 ldsP[8][16][40];   // per-wave P scratch (16x32 used)

    const int tid  = threadIdx.x;
    const int w    = tid >> 5;
    const int lane = tid & 31;
    const int half = lane >> 4;
    const int nh   = lane & 15;

    const int id  = blockIdx.x;
    const int qt  = id & 15;         // S/128 = 16 query tiles per (b,h)
    const int bh  = id >> 4;
    const int b   = bh >> 5;         // / H
    const int h   = bh & 31;
    const int hkv = h >> 2;          // H/HKV = 4

    const int    qr0  = qt * 128 + w * 16;          // query row within S
    const size_t tok0 = (size_t)b * Sc + qr0;       // global token row

    // ---- load Q fragments once (K-dim = HD = 64 -> 2 fragments) ----
    FragBF qa[2];
    {
        const __bf16* qp = q + (tok0 + nh) * QW + h * HDc;
#pragma unroll
        for (int c = 0; c < 2; ++c) {
            qa[c].h[0] = *(const v8bf*)(qp + c * 32 + half * 8);
            qa[c].h[1] = *(const v8bf*)(qp + c * 32 + 16 + half * 8);
        }
    }

    v8f acc[4];
#pragma unroll
    for (int t = 0; t < 4; ++t) acc[t] = zero8();
    float mrun[8], lrun[8];
#pragma unroll
    for (int r = 0; r < 8; ++r) { mrun[r] = -1e30f; lrun[r] = 0.f; }

    for (int kb = 0; kb < Sc; kb += 32) {
        // ---- scores S = Q K^T for 32 keys: 2 tiles x 2 K-chunks ----
        v8f sc[2];
#pragma unroll
        for (int j = 0; j < 2; ++j) {
            const __bf16* kp = k + ((size_t)b * Sc + kb + j * 16 + nh) * KW + hkv * HDc;
            FragBF kf0, kf1;
            kf0.h[0] = *(const v8bf*)(kp + half * 8);
            kf0.h[1] = *(const v8bf*)(kp + 16 + half * 8);
            kf1.h[0] = *(const v8bf*)(kp + 32 + half * 8);
            kf1.h[1] = *(const v8bf*)(kp + 48 + half * 8);
            v8f sz = zero8();
            sz = wmma_bf16(qa[0].v, kf0.v, sz);
            sz = wmma_bf16(qa[1].v, kf1.v, sz);
            sc[j] = sz;
        }

        // ---- online softmax over 32 key columns ----
#pragma unroll
        for (int r = 0; r < 8; ++r) {
            int   m_idx = half * 8 + r;
            int   qpos  = qr0 + m_idx;
            float mk0 = mask[((size_t)b * Sc + qpos) * Sc + kb + nh];
            float mk1 = mask[((size_t)b * Sc + qpos) * Sc + kb + 16 + nh];
            float s0 = sc[0][r] * scale + mk0;
            float s1 = sc[1][r] * scale + mk1;
            // row max across the 16-lane half (rows are disjoint per half)
            float v = fmaxf(s0, s1);
            v = fmaxf(v, __shfl_xor(v, 1, 32));
            v = fmaxf(v, __shfl_xor(v, 2, 32));
            v = fmaxf(v, __shfl_xor(v, 4, 32));
            v = fmaxf(v, __shfl_xor(v, 8, 32));
            float mnew  = fmaxf(mrun[r], v);
            float alpha = __expf(mrun[r] - mnew);
            mrun[r] = mnew;
            float p0 = __expf(s0 - mnew);
            float p1 = __expf(s1 - mnew);
            float rs = p0 + p1;
            rs += __shfl_xor(rs, 1, 32);
            rs += __shfl_xor(rs, 2, 32);
            rs += __shfl_xor(rs, 4, 32);
            rs += __shfl_xor(rs, 8, 32);
            lrun[r] = lrun[r] * alpha + rs;
#pragma unroll
            for (int t = 0; t < 4; ++t) acc[t][r] *= alpha;
            ldsP[w][m_idx][nh]      = (__bf16)p0;
            ldsP[w][m_idx][16 + nh] = (__bf16)p1;
        }

        // ---- reload P as A-fragment (per-wave LDS, DS ops in-order per wave)
        FragBF pa;
        pa.h[0] = *(const v8bf*)&ldsP[w][nh][half * 8];
        pa.h[1] = *(const v8bf*)&ldsP[w][nh][16 + half * 8];

        // ---- ctx += P V : 4 N-fragments of d, K-dim = 32 keys ----
#pragma unroll
        for (int t = 0; t < 4; ++t) {
            const __bf16* vp = vT + ((size_t)b * KW + hkv * HDc + t * 16 + nh) * Sc
                                  + kb + half * 8;
            FragBF vf;
            vf.h[0] = *(const v8bf*)vp;
            vf.h[1] = *(const v8bf*)(vp + 16);
            acc[t] = wmma_bf16(pa.v, vf.v, acc[t]);
        }
    }

    // ---- finalize: divide by row sums, store bf16 ctx ----
#pragma unroll
    for (int r = 0; r < 8; ++r) {
        float inv = 1.f / lrun[r];
        int   m_idx = half * 8 + r;
        size_t row = tok0 + m_idx;
#pragma unroll
        for (int t = 0; t < 4; ++t) {
            ctx[row * QW + h * HDc + t * 16 + nh] = (__bf16)(acc[t][r] * inv);
        }
    }
}

// ---------------------------------------------------------------------------
// Host launcher
// inputs: hidden[2,2048,2048] f32, mask[2,1,2048,2048] f32,
//         Wq[2048,2048], Wk[512,2048], Wv[512,2048], Wo[2048,2048] (all f32)
// output: [2,2048,2048] f32
// ---------------------------------------------------------------------------
extern "C" void kernel_launch(void* const* d_in, const int* in_sizes, int n_in,
                              void* d_out, int out_size, void* d_ws, size_t ws_size,
                              hipStream_t stream) {
    (void)in_sizes; (void)n_in; (void)out_size; (void)ws_size;
    const float* hs   = (const float*)d_in[0];
    const float* mask = (const float*)d_in[1];
    const float* Wq   = (const float*)d_in[2];
    const float* Wk   = (const float*)d_in[3];
    const float* Wv   = (const float*)d_in[4];
    const float* Wo   = (const float*)d_in[5];
    float* out = (float*)d_out;

    constexpr int B = 2, S = 2048, D = 2048, H = 32, HKV = 8, HD = 64;
    constexpr int M = B * S;            // 4096
    constexpr int NQ = H * HD;          // 2048
    constexpr int NKV = HKV * HD;       // 512

    __bf16* qb = (__bf16*)d_ws;                         // [M, NQ]
    __bf16* kb = qb + (size_t)M * NQ;                   // [M, NKV]
    __bf16* vT = kb + (size_t)M * NKV;                  // [B, NKV, S]
    __bf16* cx = vT + (size_t)B * NKV * S;              // [M, NQ]

    dim3 blk(256);
    // Q = hidden @ Wq^T  (bf16 out)
    gemm_nt_kernel<true, false, false>
        <<<dim3(NQ / 128, M / 128), blk, 0, stream>>>(hs, Wq, qb, M, NQ, D, S);
    // K = hidden @ Wk^T  (bf16 out)
    gemm_nt_kernel<true, false, false>
        <<<dim3(NKV / 128, M / 128), blk, 0, stream>>>(hs, Wk, kb, M, NKV, D, S);
    // V^T = (hidden @ Wv^T)^T per batch  (bf16 out, transposed)
    gemm_nt_kernel<true, false, true>
        <<<dim3(NKV / 128, M / 128), blk, 0, stream>>>(hs, Wv, vT, M, NKV, D, S);
    // flash attention -> ctx (bf16)
    attn_kernel<<<dim3(B * H * (S / 128)), blk, 0, stream>>>(qb, kb, vT, mask, cx);
    // out = ctx @ Wo^T  (fp32 out, A tile staged via Tensor Data Mover)
    gemm_nt_kernel<false, true, false>
        <<<dim3(D / 128, M / 128), blk, 0, stream>>>(cx, Wo, out, M, D, NQ, S);
}
